// Qwen3RoutMoeSparseMoeBlock_51625506898613
// MI455X (gfx1250) — matrix operand order repaired and checked by reading, not compiled, not run on hardware
//
#include <hip/hip_runtime.h>
#include <hip/hip_bf16.h>

#define HDIM 2048
#define IDIM 768
#define EDIM 16
#define ZCOEFF 0.01f
#define EPSV 1e-8f
#define KT1 (HDIM / 32)   // 64  K-tiles, stage 1 (K = H)
#define KT2 (IDIM / 32)   // 24  K-tiles, stage 2 (K = I)
#define CT1 (IDIM / 16)   // 48  N-tiles, stage 1
#define CT2 (HDIM / 16)   // 128 N-tiles, stage 2

typedef __attribute__((ext_vector_type(16))) __bf16 v16bf;
typedef __attribute__((ext_vector_type(8)))  float  v8f;

// ---------------- router: logits = x @ gate_w^T, plus z-loss sum of squares ----
__global__ void router_kernel(const float* __restrict__ x,
                              const float* __restrict__ gw,
                              float* __restrict__ logits,
                              float* __restrict__ zacc,
                              int N)
{
    int wid  = (blockIdx.x * blockDim.x + threadIdx.x) >> 5;
    int lane = threadIdx.x & 31;
    if (wid >= N) return;
    const float* xr = x + (size_t)wid * HDIM;
    float acc[EDIM];
#pragma unroll
    for (int e = 0; e < EDIM; ++e) acc[e] = 0.f;
    for (int k = lane; k < HDIM; k += 32) {
        float xv = xr[k];
#pragma unroll
        for (int e = 0; e < EDIM; ++e)
            acc[e] = fmaf(xv, gw[e * HDIM + k], acc[e]);
    }
#pragma unroll
    for (int e = 0; e < EDIM; ++e) {
#pragma unroll
        for (int off = 16; off > 0; off >>= 1)
            acc[e] += __shfl_xor(acc[e], off, 32);
    }
    if (lane == 0) {
        float mean = 0.f;
#pragma unroll
        for (int e = 0; e < EDIM; ++e) mean += acc[e];
        mean *= (1.f / EDIM);
        float ss = 0.f;
#pragma unroll
        for (int e = 0; e < EDIM; ++e) {
            logits[(size_t)wid * EDIM + e] = acc[e];
            float c = acc[e] - mean;
            ss = fmaf(c, c, ss);
        }
        atomicAdd(zacc, ss);
    }
}

// ---------------- z-scale + softmax + top-8 mask + renorm -> weights ----------
__global__ void weights_kernel(const float* __restrict__ logits,
                               const float* __restrict__ zacc,
                               float* __restrict__ wts, int N)
{
    int n = blockIdx.x * blockDim.x + threadIdx.x;
    if (n >= N) return;
    float z = ZCOEFF * zacc[0] / (float)(N * EDIM);
    float s = 1.f - z;
    float p[EDIM];
    float mx = -3.4e38f;
#pragma unroll
    for (int e = 0; e < EDIM; ++e) {
        p[e] = logits[(size_t)n * EDIM + e] * s;
        mx = fmaxf(mx, p[e]);
    }
    float se = 0.f;
#pragma unroll
    for (int e = 0; e < EDIM; ++e) { p[e] = __expf(p[e] - mx); se += p[e]; }
    float inv = 1.f / se;
#pragma unroll
    for (int e = 0; e < EDIM; ++e) p[e] *= inv;
    float ssum = 0.f;
    float w[EDIM];
#pragma unroll
    for (int e = 0; e < EDIM; ++e) {
        int cnt = 0;
#pragma unroll
        for (int j = 0; j < EDIM; ++j)
            cnt += (p[j] > p[e]) || (p[j] == p[e] && j < e);
        w[e] = (cnt < 8) ? p[e] : 0.f;
        ssum += w[e];
    }
    float invd = 1.f / fmaxf(ssum, EPSV);
#pragma unroll
    for (int e = 0; e < EDIM; ++e)
        wts[(size_t)n * EDIM + e] = w[e] * invd;
}

// ============================ FAST (packed) PATH ==============================
// A-fragment packing: [M/16][K/32][lane:32][16 bf16]; per lane K-order is
// K = (i/8)*16 + (lane/16)*8 + i%8 (ISA 16-bit A layout). 32B/lane contiguous.
__global__ void pack_a_kernel(const float* __restrict__ src, // [M,K] f32
                              unsigned short* __restrict__ dstraw, int M, int K)
{
    __bf16* dst = (__bf16*)dstraw;
    int t = blockIdx.x * blockDim.x + threadIdx.x;
    int ktiles = K >> 5;
    int total = (M >> 4) * ktiles * 32;
    if (t >= total) return;
    int lane = t & 31;
    int tile = t >> 5;
    int kt = tile % ktiles;
    int mt = tile / ktiles;
    int half = lane >> 4, m = lane & 15;
    const float* s = src + (size_t)(mt * 16 + m) * K + kt * 32 + half * 8;
    __bf16* d = dst + ((size_t)t << 4);
#pragma unroll
    for (int i = 0; i < 8; ++i) {
        d[i]     = (__bf16)s[i];
        d[i + 8] = (__bf16)s[i + 16];
    }
}

// B-fragment packing: nmat row-major [K,N] matrices -> per matrix
// [N/16][K/32][lane:32][16 bf16]; lane n = lane&15 holds K = (i/8)*16 + (lane/16)*8 + i%8.
__global__ void pack_b_kernel(const float* __restrict__ src,
                              unsigned short* __restrict__ dstraw,
                              int K, int N, int nmat)
{
    __bf16* dst = (__bf16*)dstraw;
    int t = blockIdx.x * blockDim.x + threadIdx.x;
    int ktiles = K >> 5, ntiles = N >> 4;
    int total = nmat * ntiles * ktiles * 32;
    if (t >= total) return;
    int lane = t & 31;
    int tile = t >> 5;
    int kt = tile % ktiles;
    int rem = tile / ktiles;
    int nt = rem % ntiles;
    int e  = rem / ntiles;
    int half = lane >> 4, n = lane & 15;
    const float* s = src + (size_t)e * K * N
                   + (size_t)(kt * 32 + half * 8) * N + nt * 16 + n;
    __bf16* d = dst + ((size_t)t << 4);
#pragma unroll
    for (int i = 0; i < 16; ++i)
        d[i] = (__bf16)s[(size_t)(i + (i & 8)) * N];
}

// Map (token, I-col) element into the stage-2 A-fragment layout of h.
__device__ __forceinline__ size_t hfrag_idx(int token, int icol)
{
    int mt = token >> 4, mr = token & 15;
    int kt = icol >> 5, kk = icol & 31;
    int half = (kk & 15) >> 3;
    int i = ((kk >> 4) << 3) + (kk & 7);
    int lane = mr + (half << 4);
    return ((((size_t)mt * KT2 + kt) * 32 + lane) << 4) + (size_t)i;
}

// Stage 1 (packed): each wave -> 32 tokens x 16 I-cols, gate+up fused.
__global__ void gateup_fast_kernel(const unsigned short* __restrict__ axraw,
                                   const unsigned short* __restrict__ bgraw,
                                   const unsigned short* __restrict__ buraw,
                                   unsigned short* __restrict__ hraw,
                                   int M)
{
    const __bf16* ax  = (const __bf16*)axraw;
    const __bf16* bgf = (const __bf16*)bgraw;
    const __bf16* buf = (const __bf16*)buraw;
    __bf16* h = (__bf16*)hraw;
    int wave = (blockIdx.x * blockDim.x + threadIdx.x) >> 5;
    int lane = threadIdx.x & 31;
    int rtiles = M >> 5;
    int rt = wave / CT1;
    int ct = wave - rt * CT1;
    if (rt >= rtiles) return;
    const __bf16* a0p = ax  + ((((size_t)(rt * 2) * KT1) * 32 + lane) << 4);
    const __bf16* a1p = a0p + ((size_t)KT1 * 32 << 4);
    const __bf16* bgp = bgf + ((((size_t)ct * KT1) * 32 + lane) << 4);
    const __bf16* bup = buf + ((((size_t)ct * KT1) * 32 + lane) << 4);
    v8f ag0 = {}, au0 = {}, ag1 = {}, au1 = {};
    for (int kt = 0; kt < KT1; ++kt) {
        v16bf a0 = *(const v16bf*)a0p;
        v16bf a1 = *(const v16bf*)a1p;
        v16bf bg = *(const v16bf*)bgp;
        v16bf bu = *(const v16bf*)bup;
        __builtin_prefetch((const void*)(a0p + 4096), 0, 1);  // ~8 K-steps ahead
        __builtin_prefetch((const void*)(bgp + 4096), 0, 1);
        __builtin_prefetch((const void*)(bup + 4096), 0, 1);
        a0p += 512; a1p += 512; bgp += 512; bup += 512;
        ag0 = __builtin_amdgcn_wmma_f32_16x16x32_bf16(false, a0, false, bg, (short)0, ag0, false, false);
        au0 = __builtin_amdgcn_wmma_f32_16x16x32_bf16(false, a0, false, bu, (short)0, au0, false, false);
        ag1 = __builtin_amdgcn_wmma_f32_16x16x32_bf16(false, a1, false, bg, (short)0, ag1, false, false);
        au1 = __builtin_amdgcn_wmma_f32_16x16x32_bf16(false, a1, false, bu, (short)0, au1, false, false);
    }
    int half = lane >> 4, n = lane & 15;
    int r0 = rt * 32, c0 = ct * 16;
#pragma unroll
    for (int r = 0; r < 8; ++r) {
        int m = half * 8 + r;
        float g0 = ag0[r];
        float v0 = (g0 / (1.f + __expf(-g0))) * au0[r];
        h[hfrag_idx(r0 + m, c0 + n)] = (__bf16)v0;
        float g1 = ag1[r];
        float v1 = (g1 / (1.f + __expf(-g1))) * au1[r];
        h[hfrag_idx(r0 + 16 + m, c0 + n)] = (__bf16)v1;
    }
}

// Stage 2 (packed): out += diag(w_e) * (h @ Wd)
__global__ void down_fast_kernel(const unsigned short* __restrict__ hraw,
                                 const unsigned short* __restrict__ bdraw,
                                 const float* __restrict__ wts,
                                 float* __restrict__ out,
                                 int M, int expert)
{
    const __bf16* hf = (const __bf16*)hraw;
    const __bf16* bd = (const __bf16*)bdraw;
    int wave = (blockIdx.x * blockDim.x + threadIdx.x) >> 5;
    int lane = threadIdx.x & 31;
    int rtiles = M >> 5;
    int rt = wave / CT2;
    int ct = wave - rt * CT2;
    if (rt >= rtiles) return;
    const __bf16* a0p = hf  + ((((size_t)(rt * 2) * KT2) * 32 + lane) << 4);
    const __bf16* a1p = a0p + ((size_t)KT2 * 32 << 4);
    const __bf16* bp  = bd  + ((((size_t)ct * KT2) * 32 + lane) << 4);
    v8f acc0 = {}, acc1 = {};
    for (int kt = 0; kt < KT2; ++kt) {
        v16bf a0 = *(const v16bf*)a0p;
        v16bf a1 = *(const v16bf*)a1p;
        v16bf b  = *(const v16bf*)bp;
        __builtin_prefetch((const void*)(bp + 4096), 0, 1);
        a0p += 512; a1p += 512; bp += 512;
        acc0 = __builtin_amdgcn_wmma_f32_16x16x32_bf16(false, a0, false, b, (short)0, acc0, false, false);
        acc1 = __builtin_amdgcn_wmma_f32_16x16x32_bf16(false, a1, false, b, (short)0, acc1, false, false);
    }
    int half = lane >> 4, n = lane & 15;
    int r0 = rt * 32, c0 = ct * 16;
#pragma unroll
    for (int r = 0; r < 8; ++r) {
        int m = half * 8 + r;
        int row0 = r0 + m;
        float w0 = wts[(size_t)row0 * EDIM + expert];
        size_t o0 = (size_t)row0 * HDIM + c0 + n;
        out[o0] += w0 * acc0[r];
        int row1 = row0 + 16;
        float w1 = wts[(size_t)row1 * EDIM + expert];
        size_t o1 = (size_t)row1 * HDIM + c0 + n;
        out[o1] += w1 * acc1[r];
    }
}

// ====================== FALLBACK PATH (small workspace) =======================
__global__ void gateup_kernel(const float* __restrict__ x,   // [cn, HDIM]
                              const float* __restrict__ wg,  // [HDIM, IDIM]
                              const float* __restrict__ wu,
                              unsigned short* __restrict__ hraw, // bf16 [cn, IDIM]
                              int cn)
{
    __bf16* h = (__bf16*)hraw;
    int wave = (blockIdx.x * blockDim.x + threadIdx.x) >> 5;
    int lane = threadIdx.x & 31;
    int rtiles = cn >> 5;
    int rt = wave / CT1;
    int ct = wave - rt * CT1;
    if (rt >= rtiles) return;
    int r0 = rt << 5, c0 = ct << 4;
    int half = lane >> 4, mn = lane & 15;
    v8f ag0 = {}, au0 = {}, ag1 = {}, au1 = {};
    for (int k = 0; k < HDIM; k += 32) {
        v16bf a0, a1, bg, bu;
        const float* xa = x + (size_t)(r0 + mn) * HDIM + k + half * 8;
        const float* xb = xa + (size_t)16 * HDIM;
#pragma unroll
        for (int i = 0; i < 8; ++i) {
            a0[i] = (__bf16)xa[i];  a0[i + 8] = (__bf16)xa[i + 16];
            a1[i] = (__bf16)xb[i];  a1[i + 8] = (__bf16)xb[i + 16];
        }
        const float* gp = wg + (size_t)(k + half * 8) * IDIM + c0 + mn;
        const float* up = wu + (size_t)(k + half * 8) * IDIM + c0 + mn;
#pragma unroll
        for (int i = 0; i < 16; ++i) {
            bg[i] = (__bf16)gp[(size_t)(i + (i & 8)) * IDIM];
            bu[i] = (__bf16)up[(size_t)(i + (i & 8)) * IDIM];
        }
        ag0 = __builtin_amdgcn_wmma_f32_16x16x32_bf16(false, a0, false, bg, (short)0, ag0, false, false);
        au0 = __builtin_amdgcn_wmma_f32_16x16x32_bf16(false, a0, false, bu, (short)0, au0, false, false);
        ag1 = __builtin_amdgcn_wmma_f32_16x16x32_bf16(false, a1, false, bg, (short)0, ag1, false, false);
        au1 = __builtin_amdgcn_wmma_f32_16x16x32_bf16(false, a1, false, bu, (short)0, au1, false, false);
    }
#pragma unroll
    for (int r = 0; r < 8; ++r) {
        int m = half * 8 + r;
        float g0 = ag0[r];
        float v0 = (g0 / (1.f + __expf(-g0))) * au0[r];
        h[(size_t)(r0 + m) * IDIM + c0 + mn] = (__bf16)v0;
        float g1 = ag1[r];
        float v1 = (g1 / (1.f + __expf(-g1))) * au1[r];
        h[(size_t)(r0 + 16 + m) * IDIM + c0 + mn] = (__bf16)v1;
    }
}

__global__ void down_kernel(const unsigned short* __restrict__ hraw, // bf16 [cn, IDIM]
                            const float* __restrict__ wd,   // [IDIM, HDIM]
                            const float* __restrict__ wts,  // [N, EDIM]
                            float* __restrict__ out,        // [N, HDIM]
                            int n0, int cn, int expert)
{
    const __bf16* h = (const __bf16*)hraw;
    int wave = (blockIdx.x * blockDim.x + threadIdx.x) >> 5;
    int lane = threadIdx.x & 31;
    int rtiles = cn >> 5;
    int rt = wave / CT2;
    int ct = wave - rt * CT2;
    if (rt >= rtiles) return;
    int r0 = rt << 5, c0 = ct << 4;
    int half = lane >> 4, mn = lane & 15;
    v8f acc0 = {}, acc1 = {};
    for (int k = 0; k < IDIM; k += 32) {
        v16bf a0, a1, b;
        const __bf16* ha = h + (size_t)(r0 + mn) * IDIM + k + half * 8;
        const __bf16* hb = ha + (size_t)16 * IDIM;
#pragma unroll
        for (int i = 0; i < 8; ++i) {
            a0[i] = ha[i];  a0[i + 8] = ha[i + 16];
            a1[i] = hb[i];  a1[i + 8] = hb[i + 16];
        }
        const float* dp = wd + (size_t)(k + half * 8) * HDIM + c0 + mn;
#pragma unroll
        for (int i = 0; i < 16; ++i) b[i] = (__bf16)dp[(size_t)(i + (i & 8)) * HDIM];
        acc0 = __builtin_amdgcn_wmma_f32_16x16x32_bf16(false, a0, false, b, (short)0, acc0, false, false);
        acc1 = __builtin_amdgcn_wmma_f32_16x16x32_bf16(false, a1, false, b, (short)0, acc1, false, false);
    }
#pragma unroll
    for (int r = 0; r < 8; ++r) {
        int m = half * 8 + r;
        int row0 = r0 + m;
        float w0 = wts[(size_t)(n0 + row0) * EDIM + expert];
        size_t o0 = (size_t)(n0 + row0) * HDIM + c0 + mn;
        out[o0] += w0 * acc0[r];
        int row1 = row0 + 16;
        float w1 = wts[(size_t)(n0 + row1) * EDIM + expert];
        size_t o1 = (size_t)(n0 + row1) * HDIM + c0 + mn;
        out[o1] += w1 * acc1[r];
    }
}

extern "C" void kernel_launch(void* const* d_in, const int* in_sizes, int n_in,
                              void* d_out, int out_size, void* d_ws, size_t ws_size,
                              hipStream_t stream)
{
    const float* x      = (const float*)d_in[0];   // [N, H]
    const float* gate_w = (const float*)d_in[1];   // [E, H]
    const float* w_gate = (const float*)d_in[2];   // [E, H, I]
    const float* w_up   = (const float*)d_in[3];   // [E, H, I]
    const float* w_down = (const float*)d_in[4];   // [E, I, H]
    float* out = (float*)d_out;                    // [N*H] then router_logits [N*E]
    const int N = in_sizes[0] / HDIM;
    float* logits = out + (size_t)N * HDIM;

    char* wsb = (char*)d_ws;
    float* zacc = (float*)wsb;
    size_t off = 256;
    float* wts = (float*)(wsb + off);
    off += (((size_t)N * EDIM * 4) + 255) & ~(size_t)255;
    size_t base = off;

    // Fast-path workspace layout (bf16 packed operands).
    size_t axb = (((size_t)N * HDIM * 2) + 255) & ~(size_t)255;
    size_t bwb = (((size_t)EDIM * HDIM * IDIM * 2) + 255) & ~(size_t)255;
    size_t hb  = (((size_t)N * IDIM * 2) + 255) & ~(size_t)255;
    size_t ax_off = base;
    size_t bg_off = ax_off + axb;
    size_t bu_off = bg_off + bwb;
    size_t bd_off = bu_off + bwb;
    size_t hf_off = bd_off + bwb;
    size_t need_fast = hf_off + hb;

    hipMemsetAsync(d_ws, 0, 256, stream);
    hipMemsetAsync(out, 0, (size_t)N * HDIM * sizeof(float), stream);

    router_kernel<<<(N + 7) / 8, 256, 0, stream>>>(x, gate_w, logits, zacc, N);
    weights_kernel<<<(N + 255) / 256, 256, 0, stream>>>(logits, zacc, wts, N);

    if (ws_size >= need_fast) {
        unsigned short* ax = (unsigned short*)(wsb + ax_off);
        unsigned short* bg = (unsigned short*)(wsb + bg_off);
        unsigned short* bu = (unsigned short*)(wsb + bu_off);
        unsigned short* bd = (unsigned short*)(wsb + bd_off);
        unsigned short* hf = (unsigned short*)(wsb + hf_off);

        int ta = (N >> 4) * (HDIM >> 5) * 32;
        pack_a_kernel<<<(ta + 255) / 256, 256, 0, stream>>>(x, ax, N, HDIM);
        int tb1 = EDIM * (IDIM >> 4) * (HDIM >> 5) * 32;
        pack_b_kernel<<<(tb1 + 255) / 256, 256, 0, stream>>>(w_gate, bg, HDIM, IDIM, EDIM);
        pack_b_kernel<<<(tb1 + 255) / 256, 256, 0, stream>>>(w_up,   bu, HDIM, IDIM, EDIM);
        int tb2 = EDIM * (HDIM >> 4) * (IDIM >> 5) * 32;
        pack_b_kernel<<<(tb2 + 255) / 256, 256, 0, stream>>>(w_down, bd, IDIM, HDIM, EDIM);

        int gw1 = (N >> 5) * CT1;
        int gw2 = (N >> 5) * CT2;
        dim3 g1((gw1 + 3) / 4), g2((gw2 + 3) / 4);
        size_t estride = (size_t)HDIM * IDIM;  // elements per packed expert matrix
        for (int e = 0; e < EDIM; ++e) {
            gateup_fast_kernel<<<g1, 128, 0, stream>>>(ax, bg + e * estride,
                                                       bu + e * estride, hf, N);
            down_fast_kernel<<<g2, 128, 0, stream>>>(hf, bd + e * estride,
                                                     wts, out, N, e);
        }
    } else {
        // Fallback: convert-in-loop kernels with h chunked to fit ws.
        unsigned short* hbuf = (unsigned short*)(wsb + base);
        size_t remain = (ws_size > base) ? (ws_size - base) : 0;
        long maxRows = (long)(remain / (IDIM * 2));
        int chunk = (int)((maxRows > (long)N) ? (long)N : maxRows);
        chunk &= ~31;
        if (chunk < 32) chunk = 32;
        for (int n0 = 0; n0 < N; n0 += chunk) {
            int cn = N - n0; if (cn > chunk) cn = chunk;
            int rt = cn >> 5;
            dim3 gub((rt * CT1 + 3) / 4), dnb((rt * CT2 + 3) / 4);
            for (int e = 0; e < EDIM; ++e) {
                gateup_kernel<<<gub, 128, 0, stream>>>(
                    x + (size_t)n0 * HDIM,
                    w_gate + (size_t)e * HDIM * IDIM,
                    w_up   + (size_t)e * HDIM * IDIM,
                    hbuf, cn);
                down_kernel<<<dnb, 128, 0, stream>>>(
                    hbuf, w_down + (size_t)e * IDIM * HDIM,
                    wts, out, n0, cn, e);
            }
        }
    }
}
